// MambaBlock_23613730193611
// MI455X (gfx1250) — matrix operand order repaired and checked by reading, not compiled, not run on hardware
//
#include <hip/hip_runtime.h>

#define BATCH 4
#define SEQ   4096
#define DM    1024
#define DI    2048
#define DSTATE 16
#define DTRANK 64
#define NTOK  (BATCH * SEQ)   // 16384

typedef __bf16 bf16_t;
typedef __attribute__((ext_vector_type(16))) __bf16 v16bf;
typedef __attribute__((ext_vector_type(8)))  float  v8f;
typedef int v4i __attribute__((ext_vector_type(4)));

union FragBf { v16bf v; uint4 q[2]; };

// --------------------------- async to-LDS support ---------------------------
#if defined(__has_builtin)
#if __has_builtin(__builtin_amdgcn_global_load_async_to_lds_b128) && \
    __has_builtin(__builtin_amdgcn_s_wait_asynccnt)
#define ASYNC_LDS 1
#endif
#endif
#ifndef ASYNC_LDS
#define ASYNC_LDS 0
#endif

// Copy 16 bytes global -> LDS. Async path uses GLOBAL_LOAD_ASYNC_TO_LDS_B128
// (ASYNCcnt); fallback is load + ds_store.
__device__ __forceinline__ void fill_b128(const bf16_t* gp, bf16_t* lp) {
#if ASYNC_LDS
    void* gv = (void*)gp;   // strip const, then addrspace+type cast
    void* lv = (void*)lp;
    __builtin_amdgcn_global_load_async_to_lds_b128(
        (__attribute__((address_space(1))) v4i*)gv,
        (__attribute__((address_space(3))) v4i*)lv, 0, 0);
#else
    *(uint4*)lp = *(const uint4*)gp;
#endif
}

__device__ __forceinline__ void wait_fill() {
#if ASYNC_LDS
    __builtin_amdgcn_s_wait_asynccnt(0);
#endif
}

// ---------------------------------------------------------------------------
// RMSNorm over d_model + cast to bf16. One block (256 thr) per token.
// ---------------------------------------------------------------------------
__global__ __launch_bounds__(256)
void k_rmsnorm_bf16(const float* __restrict__ x, bf16_t* __restrict__ u) {
    __shared__ float sred[8];
    int row = blockIdx.x;
    const float* xr = x + (size_t)row * DM;
    float v[4];
    float ss = 0.f;
#pragma unroll
    for (int i = 0; i < 4; ++i) {
        v[i] = xr[threadIdx.x + i * 256];
        ss += v[i] * v[i];
    }
#pragma unroll
    for (int o = 16; o >= 1; o >>= 1) ss += __shfl_xor(ss, o, 32);
    int wid = threadIdx.x >> 5;
    if ((threadIdx.x & 31) == 0) sred[wid] = ss;
    __syncthreads();
    if (threadIdx.x == 0) {
        float t = 0.f;
#pragma unroll
        for (int i = 0; i < 8; ++i) t += sred[i];
        sred[0] = rsqrtf(t / (float)DM + 1e-6f);
    }
    __syncthreads();
    float r = sred[0];
    bf16_t* ur = u + (size_t)row * DM;
#pragma unroll
    for (int i = 0; i < 4; ++i) ur[threadIdx.x + i * 256] = (bf16_t)(v[i] * r);
}

// ---------------------------------------------------------------------------
// Elementwise f32 -> bf16 cast (for weights).
// ---------------------------------------------------------------------------
__global__ void k_cast_bf16(const float* __restrict__ a, bf16_t* __restrict__ o, int n) {
    int i = blockIdx.x * 256 + threadIdx.x;
    if (i < n) o[i] = (bf16_t)a[i];
}

// ---------------------------------------------------------------------------
// WMMA GEMM: C[M,N] = A[M,K] (bf16, row-major) x W[N,K]^T (bf16, row-major)
// Block: 8 waves, tile 128(M) x 64(N). Double-buffered LDS B tile filled with
// async to-LDS copies; one s_wait_asynccnt + barrier per 32-K step. K%32==0.
// B rows n>=N are clamped (their product lands only in output cols n>=N,
// which are masked at the store).
// MODE 0: C = acc            MODE 1: C = softplus(acc + aux[col])
// MODE 2: C = acc + aux[idx] (residual, same shape as C)
// ---------------------------------------------------------------------------
template <int MODE>
__global__ __launch_bounds__(256)
void k_gemm_wmma(const bf16_t* __restrict__ A, const bf16_t* __restrict__ W,
                 float* __restrict__ C, int M, int N, int K,
                 const float* __restrict__ aux) {
    __shared__ bf16_t sB[2][64 * 40];       // 2 x (64 N-rows x 32 K, pad 40)
    int tid  = threadIdx.x;
    int lane = tid & 31;
    int wv   = tid >> 5;                    // 0..7 -> M sub-tile
    int m0   = blockIdx.y * 128 + wv * 16;
    int nb   = blockIdx.x * 64;
    int r = lane & 15, g = lane >> 4;

    // cooperative-fill addressing: one b128 per thread per step
    int fi = tid >> 2;                      // 0..63 tile row (N)
    int fj = (tid & 3) << 3;                // K half-offset 0,8,16,24
    int fn = nb + fi;
    if (fn >= N) fn = N - 1;                // clamp (masked at store)
    const bf16_t* wRow = W + (size_t)fn * K + fj;
    bf16_t* lp[2] = { &sB[0][fi * 40 + fj], &sB[1][fi * 40 + fj] };

    v8f acc[4] = {};
    const bf16_t* aRow = A + (size_t)(m0 + r) * K + 8 * g;

    int nsteps = K >> 5;                    // K/32
    fill_b128(wRow, lp[0]);                 // prime buffer 0

    for (int s = 0; s < nsteps; ++s) {
        int cur = s & 1;
        wait_fill();                        // my async fill landed in LDS
        __syncthreads();                    // everyone's fill landed
        if (s + 1 < nsteps)                 // prefetch next K tile into other buf
            fill_b128(wRow + (size_t)(s + 1) * 32, lp[cur ^ 1]);

        int kk = s << 5;
        // A fragment: lane r=M row, group g picks K base {0,8}; halves at +0,+16
        FragBf af;
        af.q[0] = *(const uint4*)(aRow + kk);
        af.q[1] = *(const uint4*)(aRow + kk + 16);

        // load all 4 B fragments first (distinct regs), then 4 WMMAs back-to-back
        FragBf bfr[4];
#pragma unroll
        for (int t = 0; t < 4; ++t) {
            const bf16_t* bp = &sB[cur][(t * 16 + r) * 40 + 8 * g];
            bfr[t].q[0] = *(const uint4*)(bp);
            bfr[t].q[1] = *(const uint4*)(bp + 16);
        }
#pragma unroll
        for (int t = 0; t < 4; ++t) {
            acc[t] = __builtin_amdgcn_wmma_f32_16x16x32_bf16(
                false, af.v, false, bfr[t].v, (short)0, acc[t], false, false);
        }
    }

    // C/D layout: VGPR i -> row m0 + i + 8*g, col nb + t*16 + r
#pragma unroll
    for (int t = 0; t < 4; ++t) {
        int col = nb + t * 16 + r;
        if (col >= N) continue;
        int rowb = m0 + g * 8;
#pragma unroll
        for (int i = 0; i < 8; ++i) {
            size_t idx = (size_t)(rowb + i) * N + col;
            float v = acc[t][i];
            if (MODE == 1) {
                v += aux[col];
                v = (v > 20.f) ? v : log1pf(__expf(v));
            } else if (MODE == 2) {
                v += aux[idx];
            }
            C[idx] = v;
        }
    }
}

// ---------------------------------------------------------------------------
// Causal depthwise conv1d (width 4) + bias + SiLU. Reads xc half of xz.
// Writes fp32 (for scan) and bf16 (for x_proj GEMM).
// ---------------------------------------------------------------------------
__global__ __launch_bounds__(256)
void k_conv_silu(const float* __restrict__ xz, const float* __restrict__ cw,
                 const float* __restrict__ cb, float* __restrict__ xc,
                 bf16_t* __restrict__ xcb) {
    size_t idx = (size_t)blockIdx.x * 256 + threadIdx.x;   // over NTOK*DI
    if (idx >= (size_t)NTOK * DI) return;
    int e = (int)(idx % DI);
    size_t row = idx / DI;
    int l = (int)(row % SEQ);
    float acc = cb[e];
#pragma unroll
    for (int k = 0; k < 4; ++k) {
        int dl = l - 3 + k;
        if (dl >= 0)
            acc += cw[e * 4 + k] * xz[(row - 3 + k) * (size_t)(2 * DI) + e];
    }
    float s = acc / (1.f + __expf(-acc));
    xc[idx]  = s;
    xcb[idx] = (bf16_t)s;
}

// ---------------------------------------------------------------------------
// Pack dt-rank slice of dbl [NTOK,96] cols 0..63 into bf16 [NTOK,64].
// ---------------------------------------------------------------------------
__global__ void k_pack_dtr(const float* __restrict__ dbl, bf16_t* __restrict__ dtr) {
    int i = blockIdx.x * 256 + threadIdx.x;       // over NTOK*64
    if (i >= NTOK * DTRANK) return;
    int row = i / DTRANK, c = i % DTRANK;
    dtr[i] = (bf16_t)dbl[row * 96 + c];
}

// ---------------------------------------------------------------------------
// Selective scan, state-parallel: thread = (b, e, s). Block = 16 channels.
// h' = h*exp(dt*A) + dt*x*B ;  y = sum_s h*C  (width-16 shuffle reduce)
// Fuses  y = (ys + D*x) * silu(z)  and bf16 cast. Prefetch distance 8 rows.
// ---------------------------------------------------------------------------
__global__ __launch_bounds__(256)
void k_scan(const float* __restrict__ dt, const float* __restrict__ dbl,
            const float* __restrict__ xc, const float* __restrict__ xz,
            const float* __restrict__ A_log, const float* __restrict__ Dsk,
            bf16_t* __restrict__ ybf) {
    int tid = threadIdx.x;
    int s   = tid & 15;
    int chl = tid >> 4;                         // 0..15 local channel
    int b   = blockIdx.x >> 7;                  // 128 e-blocks per batch
    int e   = ((blockIdx.x & 127) << 4) + chl;

    float a   = -__expf(A_log[e * DSTATE + s]);
    float dsk = Dsk[e];
    float h   = 0.f;
    size_t rowbase = (size_t)b * SEQ;

    // software pipeline: preload l=0
    float dte = dt[rowbase * DI + e];
    float xt  = xc[rowbase * DI + e];
    float Bs  = dbl[rowbase * 96 + 64 + s];
    float Cs  = dbl[rowbase * 96 + 80 + s];
    float zz  = xz[rowbase * (size_t)(2 * DI) + DI + e];

    for (int l = 0; l < SEQ; ++l) {
        if (l + 8 < SEQ) {                      // deepen the memory pipeline
            size_t rp = rowbase + l + 8;
            __builtin_prefetch(&dt[rp * DI + e], 0, 0);
            __builtin_prefetch(&xc[rp * DI + e], 0, 0);
            __builtin_prefetch(&xz[rp * (size_t)(2 * DI) + DI + e], 0, 0);
        }
        float ndte = 0.f, nxt = 0.f, nBs = 0.f, nCs = 0.f, nzz = 0.f;
        if (l + 1 < SEQ) {
            size_t rn = rowbase + l + 1;
            ndte = dt[rn * DI + e];
            nxt  = xc[rn * DI + e];
            nBs  = dbl[rn * 96 + 64 + s];
            nCs  = dbl[rn * 96 + 80 + s];
            nzz  = xz[rn * (size_t)(2 * DI) + DI + e];
        }
        h = h * __expf(a * dte) + (dte * xt) * Bs;
        float p = h * Cs;
        p += __shfl_xor(p, 8, 16);
        p += __shfl_xor(p, 4, 16);
        p += __shfl_xor(p, 2, 16);
        p += __shfl_xor(p, 1, 16);
        if (s == 0) {
            float y  = p + dsk * xt;
            float sg = zz / (1.f + __expf(-zz));
            ybf[(rowbase + l) * DI + e] = (bf16_t)(y * sg);
        }
        dte = ndte; xt = nxt; Bs = nBs; Cs = nCs; zz = nzz;
    }
}

// ---------------------------------------------------------------------------
extern "C" void kernel_launch(void* const* d_in, const int* in_sizes, int n_in,
                              void* d_out, int out_size, void* d_ws, size_t ws_size,
                              hipStream_t stream) {
    const float* x         = (const float*)d_in[0];
    const float* in_proj_w = (const float*)d_in[1];
    const float* conv_w    = (const float*)d_in[2];
    const float* conv_b    = (const float*)d_in[3];
    const float* x_proj_w  = (const float*)d_in[4];
    const float* dt_proj_w = (const float*)d_in[5];
    const float* dt_proj_b = (const float*)d_in[6];
    const float* A_log     = (const float*)d_in[7];
    const float* D_skip    = (const float*)d_in[8];
    const float* out_proj_w= (const float*)d_in[9];
    float* out = (float*)d_out;

    char* wsb = (char*)d_ws;
    size_t off = 0;
    auto alloc = [&](size_t bytes) -> char* {
        char* p = wsb + off;
        off += (bytes + 255) & ~(size_t)255;
        return p;
    };

    bf16_t* u_bf  = (bf16_t*)alloc((size_t)NTOK * DM * 2);
    bf16_t* w1    = (bf16_t*)alloc((size_t)(2 * DI) * DM * 2);        // in_proj
    bf16_t* w2    = (bf16_t*)alloc((size_t)96 * DI * 2);              // x_proj
    bf16_t* w3    = (bf16_t*)alloc((size_t)DI * DTRANK * 2);          // dt_proj
    bf16_t* w4    = (bf16_t*)alloc((size_t)DM * DI * 2);              // out_proj
    float*  xz    = (float*)alloc((size_t)NTOK * (2 * DI) * 4);
    float*  xc    = (float*)alloc((size_t)NTOK * DI * 4);
    bf16_t* xcb   = (bf16_t*)alloc((size_t)NTOK * DI * 2);
    float*  dbl   = (float*)alloc((size_t)NTOK * 96 * 4);
    bf16_t* dtrb  = (bf16_t*)alloc((size_t)NTOK * DTRANK * 2);
    float*  dtbuf = (float*)alloc((size_t)NTOK * DI * 4);
    bf16_t* ybf   = (bf16_t*)alloc((size_t)NTOK * DI * 2);

    // 1) RMSNorm + bf16 cast
    k_rmsnorm_bf16<<<NTOK, 256, 0, stream>>>(x, u_bf);

    // 2) weight casts to bf16
    {
        int n1 = 2 * DI * DM, n2 = 96 * DI, n3 = DI * DTRANK, n4 = DM * DI;
        k_cast_bf16<<<(n1 + 255) / 256, 256, 0, stream>>>(in_proj_w, w1, n1);
        k_cast_bf16<<<(n2 + 255) / 256, 256, 0, stream>>>(x_proj_w,  w2, n2);
        k_cast_bf16<<<(n3 + 255) / 256, 256, 0, stream>>>(dt_proj_w, w3, n3);
        k_cast_bf16<<<(n4 + 255) / 256, 256, 0, stream>>>(out_proj_w, w4, n4);
    }

    // 3) in_proj GEMM: xz = u x W1^T   [16384, 4096]
    k_gemm_wmma<0><<<dim3((2 * DI) / 64, NTOK / 128), 256, 0, stream>>>(
        u_bf, w1, xz, NTOK, 2 * DI, DM, nullptr);

    // 4) depthwise conv + SiLU
    {
        size_t n = (size_t)NTOK * DI;
        k_conv_silu<<<(unsigned)((n + 255) / 256), 256, 0, stream>>>(
            xz, conv_w, conv_b, xc, xcb);
    }

    // 5) x_proj GEMM: dbl = xc x W2^T   [16384, 96]
    k_gemm_wmma<0><<<dim3((96 + 63) / 64, NTOK / 128), 256, 0, stream>>>(
        xcb, w2, dbl, NTOK, 96, DI, nullptr);

    // 6) pack dt-rank slice to bf16
    k_pack_dtr<<<(NTOK * DTRANK + 255) / 256, 256, 0, stream>>>(dbl, dtrb);

    // 7) dt_proj GEMM + bias + softplus: dt = softplus(dtr x W3^T + b)
    k_gemm_wmma<1><<<dim3(DI / 64, NTOK / 128), 256, 0, stream>>>(
        dtrb, w3, dtbuf, NTOK, DI, DTRANK, dt_proj_b);

    // 8) selective scan (fuses skip + gate + bf16 cast)
    k_scan<<<BATCH * (DI / 16), 256, 0, stream>>>(
        dtbuf, dbl, xc, xz, A_log, D_skip, ybf);

    // 9) out_proj GEMM + residual: out = x + y x W4^T
    k_gemm_wmma<2><<<dim3(DM / 64, NTOK / 128), 256, 0, stream>>>(
        ybf, w4, out, NTOK, DM, DI, x);

    (void)in_sizes; (void)n_in; (void)out_size; (void)ws_size;
}
